// MPS_1245540516431
// MI455X (gfx1250) — matrix operand order repaired and checked by reading, compiled, not run on hardware
//
#include <hip/hip_runtime.h>
#include <math.h>

typedef __bf16 bf16_t;
typedef bf16_t v2bf  __attribute__((ext_vector_type(2)));
typedef bf16_t v8bf  __attribute__((ext_vector_type(8)));
typedef bf16_t v16bf __attribute__((ext_vector_type(16)));
typedef float  v8f   __attribute__((ext_vector_type(8)));

#if defined(__HIP_DEVICE_COMPILE__) && !__has_builtin(__builtin_amdgcn_wmma_f32_16x16x32_bf16)
#error "device pass: __builtin_amdgcn_wmma_f32_16x16x32_bf16 not available"
#endif

#define DIMB   64       // bond dimension
#define PDIM   2
#define W2     128      // PDIM*DIMB
#define NSITES 64
#define NBULK  62
#define BATCH  8192
#define STR    72       // padded LDS row stride in bf16 elements (mult of 8 -> b128-aligned)
#define WAVES  4
#define BLOCK  (WAVES * 32)
#define RPB    (WAVES * 16)   // rows per block (fallback kernel)

__device__ __forceinline__ v8f wmma_bf16(v16bf a, v16bf b, v8f c) {
#if defined(__HIP_DEVICE_COMPILE__)
    return __builtin_amdgcn_wmma_f32_16x16x32_bf16(
            false, a, false, b, (short)0, c, false, false);
#else
    return c;
#endif
}

__device__ __forceinline__ v16bf ld_frag16(const bf16_t* p0, const bf16_t* p1) {
    v8bf lo = *(const v8bf*)p0;
    v8bf hi = *(const v8bf*)p1;
    return __builtin_shufflevector(lo, hi, 0, 1, 2, 3, 4, 5, 6, 7,
                                           8, 9, 10, 11, 12, 13, 14, 15);
}

// ---------------------------------------------------------------------------
// Prep: bulk f32 [62][64][128] -> wsW bf16 [62][128(c)][64(k)]  (K-contiguous)
// ---------------------------------------------------------------------------
__global__ __launch_bounds__(256) void mps_prep_kernel(
    const float* __restrict__ bulk, bf16_t* __restrict__ wsW)
{
    int i = blockIdx.x * 256 + threadIdx.x;
    const int total = NBULK * W2 * DIMB;
    if (i >= total) return;
    int site = i >> 13;            // / (128*64)
    int c    = (i >> 6) & 127;
    int k    = i & 63;
    wsW[i] = (bf16_t)bulk[((size_t)site * DIMB + k) * W2 + c];
}

// ---------------------------------------------------------------------------
// Fast psi kernel: 1 wave / block, 16 batch rows per wave, no barriers.
// B-fragments load straight from pre-transposed bf16 weights in global (L2).
// ---------------------------------------------------------------------------
__global__ __launch_bounds__(32) void mps_psi_fast(
    const int*    __restrict__ cfg,
    const float*  __restrict__ left,
    const bf16_t* __restrict__ wsW,    // [62][128][64] bf16
    const float*  __restrict__ right,
    float*        __restrict__ accum)
{
    __shared__ __attribute__((aligned(16))) bf16_t envB[16 * STR];
    __shared__ float rS[W2];

    const int lane    = threadIdx.x;
    const int rowbase = blockIdx.x * 16;
    const int m  = lane & 15;
    const int hl = lane >> 4;

    // env0[r][k] = left[cfg[row,0]][k]
    for (int i = lane; i < 16 * DIMB; i += 32) {
        int r = i >> 6, k = i & 63;
        int s0 = cfg[(rowbase + r) * NSITES + 0];
        envB[r * STR + k] = (bf16_t)left[s0 * DIMB + k];
    }
    for (int i = lane; i < W2; i += 32) rS[i] = right[i];
    __syncthreads();   // single-wave workgroup: lowers to S_NOP, kept for clarity

    int srow = cfg[(rowbase + m) * NSITES + 1];   // spin of row m at next site

    for (int site = 0; site < NBULK; ++site) {
        const bf16_t* Wp = wsW + (size_t)site * W2 * DIMB;

        if (site + 1 < NBULK)
            __builtin_prefetch(Wp + W2 * DIMB + lane * 256, 0, 1);

        // spins for the 8 D-layout rows this lane owns
        int sv[8];
        #pragma unroll
        for (int v = 0; v < 8; ++v)
            sv[v] = __shfl(srow, v + (hl << 3));

        // A fragments (16x32 bf16), K chunks kc = 0,1 from LDS env
        v16bf a[2];
        #pragma unroll
        for (int kc = 0; kc < 2; ++kc) {
            const bf16_t* base = &envB[m * STR + kc * 32 + 8 * hl];
            a[kc] = ld_frag16(base, base + 16);
        }

        // 8 output tiles: B fragments straight from global bf16 weights
        v8f accv[8];
        #pragma unroll
        for (int t = 0; t < 8; ++t) {
            v8f acc = {0.f, 0.f, 0.f, 0.f, 0.f, 0.f, 0.f, 0.f};
            #pragma unroll
            for (int kc = 0; kc < 2; ++kc) {
                const bf16_t* bb = Wp + (t * 16 + m) * DIMB + kc * 32 + 8 * hl;
                v16bf b = ld_frag16(bb, bb + 16);
                acc = wmma_bf16(a[kc], b, acc);
            }
            accv[t] = acc;
        }

        // preload next site's spins (overlaps with writeback)
        if (site + 1 < NBULK)
            srow = cfg[(rowbase + m) * NSITES + site + 2];

        // branch-free selection writeback: row picks tiles {4s..4s+3}
        #pragma unroll
        for (int tt = 0; tt < 4; ++tt) {
            #pragma unroll
            for (int v = 0; v < 8; ++v) {
                float val = sv[v] ? accv[tt + 4][v] : accv[tt][v];
                envB[(v + (hl << 3)) * STR + tt * 16 + m] = (bf16_t)val;
            }
        }
    }

    // psi[r] = sum_k env[r][k] * right[k][s_last]
    float partial = 0.f;
    {
        int sl = cfg[(rowbase + m) * NSITES + (NSITES - 1)];
        float psi = 0.f;
        for (int k = 0; k < DIMB; ++k)
            psi += (float)envB[m * STR + k] * rS[k * 2 + sl];
        partial = (lane < 16) ? __logf(fmaxf(psi * psi, 1e-12f)) : 0.f;
    }
    for (int off = 16; off; off >>= 1)
        partial += __shfl_down(partial, off);
    if (lane == 0) atomicAdd(accum, partial);
}

// ---------------------------------------------------------------------------
// Fallback psi kernel (LDS weight staging) for small ws_size.
// ---------------------------------------------------------------------------
__global__ __launch_bounds__(BLOCK) void mps_psi_kernel(
    const int*   __restrict__ cfg,
    const float* __restrict__ left,
    const float* __restrict__ bulk,
    const float* __restrict__ right,
    float*       __restrict__ accum)
{
    __shared__ __attribute__((aligned(16))) bf16_t envB[RPB * STR];
    __shared__ __attribute__((aligned(16))) bf16_t wTB[W2 * STR];
    __shared__ __attribute__((aligned(16))) int   spinS[RPB];
    __shared__ float rS[W2];
    __shared__ float redS[WAVES];

    const int tid     = threadIdx.x;
    const int lane    = tid & 31;
    const int wave    = tid >> 5;
    const int rowbase = blockIdx.x * RPB;
    const int wrow    = wave * 16;

    for (int i = tid; i < RPB * DIMB; i += BLOCK) {
        int r = i >> 6, k = i & 63;
        int s0 = cfg[(rowbase + r) * NSITES + 0];
        envB[r * STR + k] = (bf16_t)left[s0 * DIMB + k];
    }
    if (tid < W2) rS[tid] = right[tid];
    __syncthreads();

    const int m  = lane & 15;
    const int hl = lane >> 4;

    for (int site = 0; site < NBULK; ++site) {
        const float* Bp = bulk + (size_t)site * DIMB * W2;
        if (site + 1 < NBULK)
            __builtin_prefetch(Bp + DIMB * W2 + tid * 64, 0, 1);

        for (int i = tid; i < (DIMB / 2) * (W2 / 4); i += BLOCK) {
            int kp = i >> 5;
            int c4 = (i & 31) * 4;
            float4 x0 = *(const float4*)(Bp + (2 * kp + 0) * W2 + c4);
            float4 x1 = *(const float4*)(Bp + (2 * kp + 1) * W2 + c4);
            *(v2bf*)&wTB[(c4 + 0) * STR + 2 * kp] = (v2bf){(bf16_t)x0.x, (bf16_t)x1.x};
            *(v2bf*)&wTB[(c4 + 1) * STR + 2 * kp] = (v2bf){(bf16_t)x0.y, (bf16_t)x1.y};
            *(v2bf*)&wTB[(c4 + 2) * STR + 2 * kp] = (v2bf){(bf16_t)x0.z, (bf16_t)x1.z};
            *(v2bf*)&wTB[(c4 + 3) * STR + 2 * kp] = (v2bf){(bf16_t)x0.w, (bf16_t)x1.w};
        }
        if (lane < 16)
            spinS[wrow + lane] = cfg[(rowbase + wrow + lane) * NSITES + site + 1];
        __syncthreads();

        const int4* spp = (const int4*)&spinS[wrow + (hl << 3)];
        int4 sA = spp[0], sB = spp[1];
        const int sv[8] = {sA.x, sA.y, sA.z, sA.w, sB.x, sB.y, sB.z, sB.w};

        v16bf a[2];
        #pragma unroll
        for (int kc = 0; kc < 2; ++kc) {
            const bf16_t* base = &envB[(wrow + m) * STR + kc * 32 + 8 * hl];
            a[kc] = ld_frag16(base, base + 16);
        }

        v8f accv[8];
        #pragma unroll
        for (int t = 0; t < 8; ++t) {
            v8f acc = {0.f, 0.f, 0.f, 0.f, 0.f, 0.f, 0.f, 0.f};
            #pragma unroll
            for (int kc = 0; kc < 2; ++kc) {
                const bf16_t* bb = &wTB[(t * 16 + m) * STR + kc * 32 + 8 * hl];
                v16bf b = ld_frag16(bb, bb + 16);
                acc = wmma_bf16(a[kc], b, acc);
            }
            accv[t] = acc;
        }

        #pragma unroll
        for (int tt = 0; tt < 4; ++tt) {
            #pragma unroll
            for (int v = 0; v < 8; ++v) {
                float val = sv[v] ? accv[tt + 4][v] : accv[tt][v];
                envB[(wrow + v + (hl << 3)) * STR + tt * 16 + m] = (bf16_t)val;
            }
        }
        __syncthreads();
    }

    float partial = 0.f;
    if (tid < RPB) {
        int r  = tid;
        int sl = cfg[(rowbase + r) * NSITES + (NSITES - 1)];
        float psi = 0.f;
        for (int k = 0; k < DIMB; ++k)
            psi += (float)envB[r * STR + k] * rS[k * 2 + sl];
        partial = __logf(fmaxf(psi * psi, 1e-12f));
    }
    for (int off = 16; off; off >>= 1)
        partial += __shfl_down(partial, off);
    if (lane == 0) redS[wave] = partial;
    __syncthreads();
    if (tid == 0) {
        float s = 0.f;
        for (int w = 0; w < WAVES; ++w) s += redS[w];
        atomicAdd(accum, s);
    }
}

// ---------------------------------------------------------------------------
// Kernel B: transfer-matrix log-norm (tiny; single block), final scalar output
// ---------------------------------------------------------------------------
__global__ __launch_bounds__(256) void mps_norm_kernel(
    const float* __restrict__ left,
    const float* __restrict__ bulk,
    const float* __restrict__ right,
    const float* __restrict__ accum,
    float*       __restrict__ out)
{
    __shared__ float env[DIMB * DIMB];
    __shared__ float T0[DIMB * DIMB];
    __shared__ float T1[DIMB * DIMB];
    __shared__ float red[256];
    const int tid = threadIdx.x;

    for (int i = tid; i < DIMB * DIMB; i += 256) {
        int r = i >> 6, c = i & 63;
        env[i] = left[r] * left[c] + left[DIMB + r] * left[DIMB + c];
    }
    __syncthreads();

    float log_scale = 0.f;
    for (int site = 0; site < NBULK; ++site) {
        const float* M = bulk + (size_t)site * DIMB * W2;
        for (int i = tid; i < DIMB * DIMB; i += 256) {
            int d = i >> 6, j = i & 63;
            float t0 = 0.f, t1 = 0.f;
            for (int e = 0; e < DIMB; ++e) {
                float ev = env[d * DIMB + e];
                t0 += ev * M[e * W2 + j];
                t1 += ev * M[e * W2 + 64 + j];
            }
            T0[i] = t0; T1[i] = t1;
        }
        __syncthreads();
        float maxabs = 0.f;
        for (int i = tid; i < DIMB * DIMB; i += 256) {
            int r = i >> 6, c = i & 63;
            float acc = 0.f;
            for (int d = 0; d < DIMB; ++d)
                acc += M[d * W2 + r]      * T0[d * DIMB + c]
                     + M[d * W2 + 64 + r] * T1[d * DIMB + c];
            env[i] = acc;
            maxabs = fmaxf(maxabs, fabsf(acc));
        }
        red[tid] = maxabs; __syncthreads();
        for (int s = 128; s; s >>= 1) {
            if (tid < s) red[tid] = fmaxf(red[tid], red[tid + s]);
            __syncthreads();
        }
        float scale = fmaxf(red[0], 1e-30f);
        for (int i = tid; i < DIMB * DIMB; i += 256) env[i] /= scale;
        log_scale += __logf(scale);
        __syncthreads();
    }

    float part = 0.f;
    for (int i = tid; i < DIMB * DIMB; i += 256) {
        int r = i >> 6, c = i & 63;
        part += env[i] * (right[r * 2 + 0] * right[c * 2 + 0] +
                          right[r * 2 + 1] * right[c * 2 + 1]);
    }
    red[tid] = part; __syncthreads();
    for (int s = 128; s; s >>= 1) {
        if (tid < s) red[tid] += red[tid + s];
        __syncthreads();
    }
    if (tid == 0) {
        float log_z = __logf(fmaxf(red[0], 1e-30f)) + log_scale;
        out[0] = log_z - accum[0] / (float)BATCH;
    }
}

// ---------------------------------------------------------------------------
extern "C" void kernel_launch(void* const* d_in, const int* in_sizes, int n_in,
                              void* d_out, int out_size, void* d_ws, size_t ws_size,
                              hipStream_t stream) {
    const int*   cfg   = (const int*)d_in[0];
    const float* left  = (const float*)d_in[1];
    const float* bulk  = (const float*)d_in[2];
    const float* right = (const float*)d_in[3];
    float* accum = (float*)d_ws;

    (void)hipMemsetAsync(accum, 0, sizeof(float), stream);

    const size_t wneed = 256 + (size_t)NBULK * W2 * DIMB * sizeof(bf16_t);
    if (ws_size >= wneed) {
        bf16_t* wsW = (bf16_t*)((char*)d_ws + 256);
        const int total  = NBULK * W2 * DIMB;
        mps_prep_kernel<<<(total + 255) / 256, 256, 0, stream>>>(bulk, wsW);
        mps_psi_fast<<<BATCH / 16, 32, 0, stream>>>(cfg, left, wsW, right, accum);
    } else {
        mps_psi_kernel<<<BATCH / RPB, BLOCK, 0, stream>>>(cfg, left, bulk, right, accum);
    }
    mps_norm_kernel<<<1, 256, 0, stream>>>(left, bulk, right, accum, (float*)d_out);
}
